// GraphTripletGCN_18631568130412
// MI455X (gfx1250) — compile-verified, matrix operand order checked by reading, not run on hardware
//
#include <hip/hip_runtime.h>
#include <hip/hip_bf16.h>
#include <math.h>

#define NN      50000      // N_NODES
#define NR      100000     // N_RELS
#define NT      100000     // N_TRIPLES
#define DIN     512
#define HD      768
#define NLAYER  3
#define BIMG    16
#define EPSF    1e-5f

typedef __bf16 bf16;
typedef __attribute__((ext_vector_type(8)))  bf16  v8bf;
typedef __attribute__((ext_vector_type(16))) bf16  v16bf;
typedef __attribute__((ext_vector_type(8)))  float v8f;

// ------------------------------------------------------------------ utility
__global__ void k_zero(float* __restrict__ p, long long n) {
    long long i = (long long)blockIdx.x * blockDim.x + threadIdx.x;
    long long s = (long long)gridDim.x * blockDim.x;
    for (; i < n; i += s) p[i] = 0.0f;
}

__global__ void k_cast_bf16(const float* __restrict__ src, bf16* __restrict__ dst, long long n) {
    long long i = (long long)blockIdx.x * blockDim.x + threadIdx.x;
    long long s = (long long)gridDim.x * blockDim.x;
    for (; i < n; i += s) dst[i] = (bf16)src[i];
}

// dst[r*dst_ld + c] = (bf16) src[r*src_ld + c]
__global__ void k_cast_bf16_strided(const float* __restrict__ src, int src_ld,
                                    bf16* __restrict__ dst, int dst_ld,
                                    long long rows, int cols) {
    long long total = rows * (long long)cols;
    long long i = (long long)blockIdx.x * blockDim.x + threadIdx.x;
    long long s = (long long)gridDim.x * blockDim.x;
    for (; i < total; i += s) {
        long long r = i / cols;
        int       c = (int)(i - r * cols);
        dst[r * (long long)dst_ld + c] = (bf16)src[r * (long long)src_ld + c];
    }
}

// ------------------------------------------------------------------ WMMA GEMM
// C[M,N] = A[M,K](bf16) @ W[N,K](bf16)^T + bias.  fuse_silu: C = resid + silu(acc+bias)
// Block tile 128x128, 8 waves (4 row-bands x 2 col-bands), wave tile 32x64 =
// 2x4 v_wmma_f32_16x16x32_bf16 per K-step (A-fragment reused across 4 B tiles).
#define BM 128
#define BN 128
#define LDS_LD 40   // halves; row pitch 80B = 16B-aligned, staggers banks

__global__ __launch_bounds__(256) void k_gemm_wmma(
    const bf16*  __restrict__ A,
    const bf16*  __restrict__ W,
    const float* __restrict__ bias,
    const float* __restrict__ resid,
    float*       __restrict__ C,
    int M, int N, int K, int fuse_silu)
{
    __shared__ bf16 As[BM][LDS_LD];
    __shared__ bf16 Bs[BN][LDS_LD];

    const int tid  = threadIdx.x;
    const int lane = tid & 31;
    const int wave = tid >> 5;      // 0..7
    const int wr   = wave >> 1;     // 0..3 : 32-row band
    const int wc   = wave & 1;      // 0..1 : 64-col band
    const int ln   = lane & 15;
    const int kh   = (lane >> 4) * 8;   // ISA 16-bit A/B fragment K-offset

    const int m0 = blockIdx.y * BM;
    const int n0 = blockIdx.x * BN;

    v8f acc[2][4] = {};

    // staging assignments: A and B are both 128 rows x 32 halves -> 2 uint4/thread
    const int  a_r  = tid >> 1;            // 0..127
    const int  a_h  = (tid & 1) * 16;      // half-offset 0 / 16
    const bool a_ok = (m0 + a_r) < M;
    const long long a_row = (long long)(m0 + a_r);
    const long long b_row = (long long)(n0 + a_r);  // N=768 divisible by 128: in range

    for (int k0 = 0; k0 < K; k0 += 32) {
        // ---- stage A (128x32) : zero-fill M edge
        uint4 av0 = {0,0,0,0}, av1 = {0,0,0,0};
        if (a_ok) {
            const uint4* ap = (const uint4*)(A + a_row * K + k0 + a_h);
            av0 = ap[0];
            av1 = ap[1];
            __builtin_prefetch(A + a_row * K + k0 + 32, 0, 1);   // next K tile
        }
        *(uint4*)&As[a_r][a_h]     = av0;
        *(uint4*)&As[a_r][a_h + 8] = av1;
        // ---- stage B (128x32)
        {
            const uint4* bp = (const uint4*)(W + b_row * K + k0 + a_h);
            *(uint4*)&Bs[a_r][a_h]     = bp[0];
            *(uint4*)&Bs[a_r][a_h + 8] = bp[1];
        }
        __syncthreads();

        // ---- fragments from LDS (ds_load_b128 x2 each), ISA 16-bit layout
        v16bf afrag[2], bfrag[4];
        #pragma unroll
        for (int mt = 0; mt < 2; ++mt) {
            const int r = wr * 32 + mt * 16 + ln;
            v8bf lo = *(const v8bf*)&As[r][kh];
            v8bf hi = *(const v8bf*)&As[r][kh + 16];
            afrag[mt] = __builtin_shufflevector(lo, hi, 0,1,2,3,4,5,6,7,8,9,10,11,12,13,14,15);
        }
        #pragma unroll
        for (int nt = 0; nt < 4; ++nt) {
            const int r = wc * 64 + nt * 16 + ln;
            v8bf lo = *(const v8bf*)&Bs[r][kh];
            v8bf hi = *(const v8bf*)&Bs[r][kh + 16];
            bfrag[nt] = __builtin_shufflevector(lo, hi, 0,1,2,3,4,5,6,7,8,9,10,11,12,13,14,15);
        }

        #pragma unroll
        for (int mt = 0; mt < 2; ++mt)
            #pragma unroll
            for (int nt = 0; nt < 4; ++nt)
                acc[mt][nt] = __builtin_amdgcn_wmma_f32_16x16x32_bf16(
                    false, afrag[mt], false, bfrag[nt],
                    (short)0, acc[mt][nt], false, false);
        __syncthreads();
    }

    // ---- fused epilogue: bias (+ silu + residual), per ISA C/D layout
    const int mrow = (lane >> 4) * 8;
    #pragma unroll
    for (int mt = 0; mt < 2; ++mt) {
        #pragma unroll
        for (int nt = 0; nt < 4; ++nt) {
            const int   n_g = n0 + wc * 64 + nt * 16 + ln;
            const float bv  = bias[n_g];
            #pragma unroll
            for (int r = 0; r < 8; ++r) {
                const int m_g = m0 + wr * 32 + mt * 16 + mrow + r;
                if (m_g < M) {
                    float v = acc[mt][nt][r] + bv;
                    const size_t idx = (size_t)m_g * N + n_g;
                    if (fuse_silu) {
                        v = v / (1.0f + __expf(-v));   // silu
                        v = resid[idx] + v;            // residual
                    }
                    C[idx] = v;
                }
            }
        }
    }
}

// ------------------------------------------------------------------ message passing
// per (triple, 4-channel): agg[obj]+= ns+rel ; agg[subj]+= no+rel
__global__ __launch_bounds__(256) void k_message(
    const int*   __restrict__ triples,
    const float* __restrict__ node_states,
    const float* __restrict__ rel_states,
    float*       __restrict__ agg)
{
    const int CH = HD / 4; // 192
    long long gid = (long long)blockIdx.x * blockDim.x + threadIdx.x;
    if (gid >= (long long)NT * CH) return;
    const int t  = (int)(gid / CH);
    const int c4 = (int)(gid % CH) * 4;
    const int s  = triples[3 * t + 0];
    const int rl = triples[3 * t + 1];
    const int o  = triples[3 * t + 2];
    const float4 rg = *(const float4*)(rel_states  + (size_t)rl * HD + c4);
    const float4 ns = *(const float4*)(node_states + (size_t)s  * HD + c4);
    const float4 no = *(const float4*)(node_states + (size_t)o  * HD + c4);
    float* ao = agg + (size_t)o * HD + c4;
    float* as = agg + (size_t)s * HD + c4;
    atomicAdd(ao + 0, ns.x + rg.x); atomicAdd(ao + 1, ns.y + rg.y);
    atomicAdd(ao + 2, ns.z + rg.z); atomicAdd(ao + 3, ns.w + rg.w);
    atomicAdd(as + 0, no.x + rg.x); atomicAdd(as + 1, no.y + rg.y);
    atomicAdd(as + 2, no.z + rg.z); atomicAdd(as + 3, no.w + rg.w);
}

// ------------------------------------------------------------------ layernorm (block per row)
__global__ __launch_bounds__(256) void k_layernorm(
    const float* __restrict__ x, const float* __restrict__ g,
    const float* __restrict__ b, float* __restrict__ out, int rows)
{
    const int r = blockIdx.x;
    if (r >= rows) return;
    const float* xr = x + (size_t)r * HD;
    float vals[3], lsum = 0.f, lsq = 0.f;
    #pragma unroll
    for (int i = 0; i < 3; ++i) {
        const float v = xr[threadIdx.x + i * 256];
        vals[i] = v; lsum += v; lsq += v * v;
    }
    __shared__ float s1[256], s2[256];
    s1[threadIdx.x] = lsum; s2[threadIdx.x] = lsq;
    __syncthreads();
    for (int off = 128; off > 0; off >>= 1) {
        if (threadIdx.x < off) {
            s1[threadIdx.x] += s1[threadIdx.x + off];
            s2[threadIdx.x] += s2[threadIdx.x + off];
        }
        __syncthreads();
    }
    const float mu  = s1[0] * (1.0f / HD);
    const float var = s2[0] * (1.0f / HD) - mu * mu;
    const float inv = rsqrtf(var + EPSF);
    float* orow = out + (size_t)r * HD;
    #pragma unroll
    for (int i = 0; i < 3; ++i) {
        const int c = threadIdx.x + i * 256;
        orow[c] = (vals[i] - mu) * inv * g[c] + b[c];
    }
}

// ------------------------------------------------------------------ pooling: LDS partials
__global__ __launch_bounds__(256) void k_pool(
    const float* __restrict__ x, const int* __restrict__ img,
    float* __restrict__ gsum, float* __restrict__ gcnt)
{
    __shared__ float s_sum[BIMG * HD];   // 49152 B of the 320 KB WGP LDS
    __shared__ float s_cnt[BIMG];
    for (int i = threadIdx.x; i < BIMG * HD; i += 256) s_sum[i] = 0.f;
    if (threadIdx.x < BIMG) s_cnt[threadIdx.x] = 0.f;
    __syncthreads();
    const int node0 = blockIdx.x * 64;
    for (int i = 0; i < 64; ++i) {
        const int n = node0 + i;
        if (n >= NN) break;
        const int im = img[n];
        const float* xr = x + (size_t)n * HD;
        for (int c = threadIdx.x; c < HD; c += 256)  // (im,c) unique per thread
            s_sum[im * HD + c] += xr[c];
        if (threadIdx.x == 0) s_cnt[im] += 1.f;
    }
    __syncthreads();
    for (int i = threadIdx.x; i < BIMG * HD; i += 256) atomicAdd(&gsum[i], s_sum[i]);
    if (threadIdx.x < BIMG) atomicAdd(&gcnt[threadIdx.x], s_cnt[threadIdx.x]);
}

__global__ void k_pool_final(const float* __restrict__ gsum,
                             const float* __restrict__ gcnt,
                             float* __restrict__ out)
{
    const int i = blockIdx.x * blockDim.x + threadIdx.x;
    if (i < BIMG * HD) out[i] = gsum[i] / fmaxf(gcnt[i / HD], 1.0f);
}

// ------------------------------------------------------------------ driver
extern "C" void kernel_launch(void* const* d_in, const int* in_sizes, int n_in,
                              void* d_out, int out_size, void* d_ws, size_t ws_size,
                              hipStream_t stream) {
    (void)in_sizes; (void)n_in; (void)out_size; (void)ws_size;

    const float* node_feats = (const float*)d_in[0];
    const float* rel_feats  = (const float*)d_in[1];
    const int*   triples    = (const int*)  d_in[2];
    const int*   obj_to_img = (const int*)  d_in[3];
    const float* node_in_w  = (const float*)d_in[4];
    const float* node_in_b  = (const float*)d_in[5];
    const float* rel_in_w   = (const float*)d_in[6];
    const float* rel_in_b   = (const float*)d_in[7];
    const float* proj_w     = (const float*)d_in[8];
    const float* proj_b     = (const float*)d_in[9];
    const float* ln_node_g  = (const float*)d_in[10];
    const float* ln_node_b  = (const float*)d_in[11];
    // d_in[12..13] = ln_rel_g/b : rel layernorm result is not returned -> dead, skipped.

    // ---- workspace carve-out
    char* ws = (char*)d_ws;
    size_t off = 0;
    auto carve = [&](size_t bytes) -> void* {
        void* p = ws + off;
        off = (off + bytes + 255) & ~(size_t)255;
        return p;
    };
    float* node_states = (float*)carve((size_t)NN * HD * 4);
    float* rel_states  = (float*)carve((size_t)NR * HD * 4);
    float* agg         = (float*)carve((size_t)NN * HD * 4);
    bf16*  Xcat        = (bf16*) carve((size_t)NN * (2 * HD) * 2);
    bf16*  bfA_node    = (bf16*) carve((size_t)NN * DIN * 2);
    bf16*  bfA_rel     = (bf16*) carve((size_t)NR * DIN * 2);
    bf16*  bfW_node    = (bf16*) carve((size_t)HD * DIN * 2);
    bf16*  bfW_rel     = (bf16*) carve((size_t)HD * DIN * 2);
    bf16*  bfW_proj    = (bf16*) carve((size_t)NLAYER * HD * (2 * HD) * 2);
    float* pool_sum    = (float*)carve((size_t)BIMG * HD * 4);
    float* pool_cnt    = (float*)carve((size_t)BIMG * 4);

    float* out_nodes  = (float*)d_out;
    float* out_global = out_nodes + (size_t)NN * HD;

    // ---- cast inputs / weights to bf16 for WMMA
    k_cast_bf16<<<4096, 256, 0, stream>>>(node_feats, bfA_node, (long long)NN * DIN);
    k_cast_bf16<<<4096, 256, 0, stream>>>(rel_feats,  bfA_rel,  (long long)NR * DIN);
    k_cast_bf16<<<1024, 256, 0, stream>>>(node_in_w,  bfW_node, (long long)HD * DIN);
    k_cast_bf16<<<1024, 256, 0, stream>>>(rel_in_w,   bfW_rel,  (long long)HD * DIN);
    k_cast_bf16<<<2048, 256, 0, stream>>>(proj_w,     bfW_proj, (long long)NLAYER * HD * 2 * HD);

    // ---- input projections (WMMA, bias fused)
    k_gemm_wmma<<<dim3(HD / BN, (NN + BM - 1) / BM), 256, 0, stream>>>(
        bfA_node, bfW_node, node_in_b, nullptr, node_states, NN, HD, DIN, 0);
    k_gemm_wmma<<<dim3(HD / BN, (NR + BM - 1) / BM), 256, 0, stream>>>(
        bfA_rel, bfW_rel, rel_in_b, nullptr, rel_states, NR, HD, DIN, 0);

    // ---- 3 GCN layers
    const long long msg_threads = (long long)NT * (HD / 4);
    for (int l = 0; l < NLAYER; ++l) {
        k_zero<<<4096, 256, 0, stream>>>(agg, (long long)NN * HD);
        k_message<<<(unsigned)((msg_threads + 255) / 256), 256, 0, stream>>>(
            triples, node_states, rel_states, agg);
        // build concat([node_states, agg]) in bf16
        k_cast_bf16_strided<<<4096, 256, 0, stream>>>(node_states, HD, Xcat,      2 * HD, NN, HD);
        k_cast_bf16_strided<<<4096, 256, 0, stream>>>(agg,         HD, Xcat + HD, 2 * HD, NN, HD);
        // projection with fused bias + silu + residual (in-place update of node_states)
        k_gemm_wmma<<<dim3(HD / BN, (NN + BM - 1) / BM), 256, 0, stream>>>(
            Xcat, bfW_proj + (size_t)l * HD * 2 * HD, proj_b + (size_t)l * HD,
            node_states, node_states, NN, HD, 2 * HD, 1);
    }

    // ---- layernorm -> output, then segment-mean pooling
    k_layernorm<<<NN, 256, 0, stream>>>(node_states, ln_node_g, ln_node_b, out_nodes, NN);
    k_zero<<<64, 256, 0, stream>>>(pool_sum, (long long)BIMG * HD);
    k_zero<<<1, 64, 0, stream>>>(pool_cnt, BIMG);
    k_pool<<<(NN + 63) / 64, 256, 0, stream>>>(out_nodes, obj_to_img, pool_sum, pool_cnt);
    k_pool_final<<<(BIMG * HD + 255) / 256, 256, 0, stream>>>(pool_sum, pool_cnt, out_global);
}
